// GATv2GNN_1357209666174
// MI455X (gfx1250) — matrix operand order repaired and checked by reading, not compiled, not run on hardware
//
#include <hip/hip_runtime.h>
#include <hip/hip_bf16.h>

// ---------------------------------------------------------------------------
// GATv2 GNN, 8 hops, H=64, for MI455X (gfx1250, wave32, WMMA).
// GEMMs run on v_wmma_f32_16x16x32_f16 (f16 operands, f32 accumulate).
// Segment softmax via L2 atomics (encoded-uint float max, f32 add).
// ---------------------------------------------------------------------------

typedef __attribute__((ext_vector_type(16))) _Float16 v16h;
typedef __attribute__((ext_vector_type(8)))  _Float16 v8h;
typedef __attribute__((ext_vector_type(4)))  _Float16 v4h;
typedef __attribute__((ext_vector_type(8)))  float    v8f;

union AFrag { v16h v; v8h h[2]; };

#define G_HOPS 8
#define G_H    64
#define LEAKY  0.2f

// Monotone order-preserving encode of f32 into u32 so atomicMax works.
__device__ __forceinline__ unsigned enc_f32(float f) {
  unsigned u = __float_as_uint(f);
  return (u & 0x80000000u) ? ~u : (u | 0x80000000u);
}
__device__ __forceinline__ float dec_f32(unsigned k) {
  return __uint_as_float((k & 0x80000000u) ? (k ^ 0x80000000u) : ~k);
}

__device__ __forceinline__ void atomic_fadd(float* p, float v) {
  __hip_atomic_fetch_add(p, v, __ATOMIC_RELAXED, __HIP_MEMORY_SCOPE_AGENT);
}

// ---------------------------------------------------------------------------
__global__ void cvt_f16_kernel(const float* __restrict__ src,
                               _Float16* __restrict__ dst, int n) {
  int i = blockIdx.x * blockDim.x + threadIdx.x;
  if (i < n) dst[i] = (_Float16)src[i];
}

__global__ void zero_u32_kernel(unsigned* __restrict__ p, int n) {
  int i = blockIdx.x * blockDim.x + threadIdx.x;
  if (i < n) p[i] = 0u;
}

// ---------------------------------------------------------------------------
// q = h @ Wq ; kn = h @ Wk   — one wave per 16-row tile, both outputs fused.
__global__ void node_gemm_kernel(const _Float16* __restrict__ h16,
                                 const _Float16* __restrict__ wq16,
                                 const _Float16* __restrict__ wk16,
                                 float* __restrict__ q,
                                 float* __restrict__ kn,
                                 int ntiles, int N) {
  int wave = (int)((blockIdx.x * blockDim.x + threadIdx.x) >> 5);
  int lane = threadIdx.x & 31;
  if (wave >= ntiles) return;                 // wave-uniform: EXEC stays full
  int hi = lane >> 4;                          // half-wave select
  int m  = lane & 15;                          // A row within tile

  const _Float16* arow = h16 + (size_t)(wave * 16 + m) * G_H;
  v8f accq[4] = {};
  v8f acck[4] = {};

  #pragma unroll
  for (int kc = 0; kc < 2; ++kc) {
    int kb = kc * 32;
    AFrag a;
    a.h[0] = *(const v8h*)(arow + kb + hi * 8);
    a.h[1] = *(const v8h*)(arow + kb + hi * 8 + 16);
    const _Float16* bq = wq16 + (size_t)kb * G_H + m;
    const _Float16* bk = wk16 + (size_t)kb * G_H + m;
    #pragma unroll
    for (int nt = 0; nt < 4; ++nt) {
      v16h bqv, bkv;
      #pragma unroll
      for (int s = 0; s < 16; ++s) {
        int k = (s < 8) ? (hi * 8 + s) : (16 + hi * 8 + (s - 8));
        bqv[s] = bq[(size_t)k * G_H + nt * 16];
        bkv[s] = bk[(size_t)k * G_H + nt * 16];
      }
      accq[nt] = __builtin_amdgcn_wmma_f32_16x16x32_f16(
          false, a.v, false, bqv, (short)0, accq[nt], false, false);
      acck[nt] = __builtin_amdgcn_wmma_f32_16x16x32_f16(
          false, a.v, false, bkv, (short)0, acck[nt], false, false);
    }
  }

  int colbase = lane & 15;
  int rowbase = wave * 16 + hi * 8;
  #pragma unroll
  for (int nt = 0; nt < 4; ++nt) {
    #pragma unroll
    for (int r = 0; r < 8; ++r) {
      int row = rowbase + r;
      if (row < N) {
        size_t off = (size_t)row * G_H + nt * 16 + colbase;
        q[off]  = accq[nt][r];
        kn[off] = acck[nt][r];
      }
    }
  }
}

// ---------------------------------------------------------------------------
// ke = e @ We ; v = kn[snd]+ke ; feat = q[rcv]+v+b ; logit = leaky(feat)·a
// Stores v (f16) and logits, atomic-max m[rcv].  One wave per 16-edge tile.
__global__ void edge_logits_kernel(const _Float16* __restrict__ e16,
                                   const _Float16* __restrict__ we16,
                                   const int* __restrict__ snd,
                                   const int* __restrict__ rcv,
                                   const float* __restrict__ q,
                                   const float* __restrict__ kn,
                                   const float* __restrict__ bias,
                                   const float* __restrict__ att,
                                   _Float16* __restrict__ v16,
                                   float* __restrict__ logits,
                                   unsigned* __restrict__ m_enc,
                                   int etiles, int E) {
  int wave = (int)((blockIdx.x * blockDim.x + threadIdx.x) >> 5);
  int lane = threadIdx.x & 31;
  if (wave >= etiles) return;                 // wave-uniform
  int hi = lane >> 4;
  int m  = lane & 15;

  const _Float16* arow = e16 + (size_t)(wave * 16 + m) * G_H;
  v8f acc[4] = {};

  #pragma unroll
  for (int kc = 0; kc < 2; ++kc) {
    int kb = kc * 32;
    AFrag a;
    a.h[0] = *(const v8h*)(arow + kb + hi * 8);
    a.h[1] = *(const v8h*)(arow + kb + hi * 8 + 16);
    const _Float16* bw = we16 + (size_t)kb * G_H + m;
    #pragma unroll
    for (int nt = 0; nt < 4; ++nt) {
      v16h bv;
      #pragma unroll
      for (int s = 0; s < 16; ++s) {
        int k = (s < 8) ? (hi * 8 + s) : (16 + hi * 8 + (s - 8));
        bv[s] = bw[(size_t)k * G_H + nt * 16];
      }
      acc[nt] = __builtin_amdgcn_wmma_f32_16x16x32_f16(
          false, a.v, false, bv, (short)0, acc[nt], false, false);
    }
  }

  // C-tile: VGPR r -> edge row (hi*8 + r); lane&15 -> column within N-tile.
  int ebase = wave * 16 + hi * 8;
  int s_idx[8], r_idx[8];
  #pragma unroll
  for (int r = 0; r < 8; ++r) {
    int eid = ebase + r;
    bool ok = eid < E;
    s_idx[r] = ok ? snd[eid] : 0;
    r_idx[r] = ok ? rcv[eid] : 0;
  }

  float partial[8] = {0.f, 0.f, 0.f, 0.f, 0.f, 0.f, 0.f, 0.f};
  #pragma unroll
  for (int nt = 0; nt < 4; ++nt) {
    int col = (lane & 15) + nt * 16;
    float bv = bias[col];
    float av = att[col];
    #pragma unroll
    for (int r = 0; r < 8; ++r) {
      int eid = ebase + r;
      float vv = acc[nt][r] + kn[(size_t)s_idx[r] * G_H + col];
      float ft = vv + q[(size_t)r_idx[r] * G_H + col] + bv;
      float act = ft > 0.f ? ft : LEAKY * ft;
      partial[r] += act * av;
      if (eid < E) v16[(size_t)eid * G_H + col] = (_Float16)vv;
    }
  }

  // Reduce each row's dot-product across the 16 lanes of its half-wave.
  #pragma unroll
  for (int r = 0; r < 8; ++r) {
    float p = partial[r];
    p += __shfl_xor(p, 1);
    p += __shfl_xor(p, 2);
    p += __shfl_xor(p, 4);
    p += __shfl_xor(p, 8);
    partial[r] = p;
  }

  if ((lane & 15) == 0) {
    #pragma unroll
    for (int r = 0; r < 8; ++r) {
      int eid = ebase + r;
      if (eid < E) {
        logits[eid] = partial[r];
        atomicMax(&m_enc[r_idx[r]], enc_f32(partial[r]));
      }
    }
  }
}

// ---------------------------------------------------------------------------
__global__ void edge_exp_kernel(const float* __restrict__ logits,
                                const int* __restrict__ rcv,
                                const unsigned* __restrict__ m_enc,
                                float* __restrict__ ex,
                                float* __restrict__ denom, int E) {
  int i = blockIdx.x * blockDim.x + threadIdx.x;
  if (i >= E) return;
  int r = rcv[i];
  float x = __expf(logits[i] - dec_f32(m_enc[r]));
  ex[i] = x;
  atomic_fadd(&denom[r], x);
}

// ---------------------------------------------------------------------------
// pooled[rcv] += alpha * v   — 4 columns per thread, f32 L2 atomics.
__global__ void edge_pool_kernel(const float* __restrict__ ex,
                                 const float* __restrict__ denom,
                                 const int* __restrict__ rcv,
                                 const _Float16* __restrict__ v16,
                                 float* __restrict__ pooled, int E) {
  int tid = blockIdx.x * blockDim.x + threadIdx.x;
  int e = tid >> 4;
  if (e >= E) return;
  int c4 = (tid & 15) * 4;
  int rv = rcv[e];
  float alpha = ex[e] / denom[rv];
  v4h vv = *(const v4h*)(v16 + (size_t)e * G_H + c4);
  #pragma unroll
  for (int j = 0; j < 4; ++j)
    atomic_fadd(&pooled[(size_t)rv * G_H + c4 + j], alpha * (float)vv[j]);
}

// ---------------------------------------------------------------------------
// h = relu(pooled) -> f16 for next hop; re-zero segment buffers; on the last
// hop, LDS-reduce and atomically accumulate the per-column node sum.
__global__ void node_update_kernel(float* __restrict__ pooled,
                                   _Float16* __restrict__ h16,
                                   unsigned* __restrict__ m_enc,
                                   float* __restrict__ denom,
                                   float* __restrict__ ctx,
                                   int N, int last) {
  __shared__ float sred[256];
  int tid = blockIdx.x * blockDim.x + threadIdx.x;
  int total = N * G_H;
  float hv = 0.f;
  if (tid < total) {
    float p = pooled[tid];
    hv = p > 0.f ? p : 0.f;
    h16[tid] = (_Float16)hv;
    pooled[tid] = 0.f;
    int c = tid & (G_H - 1);
    if (c == 0) {
      int n = tid >> 6;
      m_enc[n] = 0u;
      denom[n] = 0.f;
    }
  }
  if (last) {
    sred[threadIdx.x] = hv;
    __syncthreads();
    if (threadIdx.x < G_H) {
      float s = sred[threadIdx.x] + sred[threadIdx.x + 64] +
                sred[threadIdx.x + 128] + sred[threadIdx.x + 192];
      atomic_fadd(&ctx[threadIdx.x], s);
    }
  }
}

__global__ void finalize_kernel(const float* __restrict__ ctx,
                                float* __restrict__ out, float inv_n) {
  int i = threadIdx.x;
  if (i < G_H) out[i] = ctx[i] * inv_n;
}

// ---------------------------------------------------------------------------
extern "C" void kernel_launch(void* const* d_in, const int* in_sizes, int n_in,
                              void* d_out, int out_size, void* d_ws, size_t ws_size,
                              hipStream_t stream) {
  const float* node_feats = (const float*)d_in[0];
  const float* edge_feats = (const float*)d_in[1];
  const int*   senders    = (const int*)d_in[2];
  const int*   receivers  = (const int*)d_in[3];
  const float* Wq         = (const float*)d_in[4];
  const float* Wk         = (const float*)d_in[5];
  const float* We         = (const float*)d_in[6];
  const float* bias       = (const float*)d_in[7];
  const float* att        = (const float*)d_in[8];
  float* out = (float*)d_out;

  const int N = in_sizes[0] / G_H;
  const int E = in_sizes[2];
  const int ntiles = (N + 15) / 16;
  const int etiles = (E + 15) / 16;
  const int WHH = G_HOPS * G_H * G_H;

  // ---- carve scratch from d_ws -------------------------------------------
  size_t off = 0;
  auto alloc = [&](size_t bytes) -> void* {
    void* p = (char*)d_ws + off;
    off += (bytes + 255) & ~(size_t)255;
    return p;
  };
  _Float16* wq16   = (_Float16*)alloc((size_t)WHH * 2);
  _Float16* wk16   = (_Float16*)alloc((size_t)WHH * 2);
  _Float16* we16   = (_Float16*)alloc((size_t)WHH * 2);
  _Float16* e16    = (_Float16*)alloc((size_t)etiles * 16 * G_H * 2);
  _Float16* h16    = (_Float16*)alloc((size_t)ntiles * 16 * G_H * 2);
  float*    q      = (float*)alloc((size_t)ntiles * 16 * G_H * 4);
  float*    kn     = (float*)alloc((size_t)ntiles * 16 * G_H * 4);
  _Float16* v16    = (_Float16*)alloc((size_t)etiles * 16 * G_H * 2);
  float*    logits = (float*)alloc((size_t)E * 4);
  float*    ex     = (float*)alloc((size_t)E * 4);
  unsigned* m_enc  = (unsigned*)alloc((size_t)N * 4);
  float*    denom  = (float*)alloc((size_t)N * 4);
  float*    pooled = (float*)alloc((size_t)N * G_H * 4);
  float*    ctx    = (float*)alloc((size_t)G_H * 4);

  const int BT = 256;
  auto blks = [&](int n) { return (unsigned)((n + BT - 1) / BT); };

  // ---- one-time (per launch) prep ----------------------------------------
  cvt_f16_kernel<<<blks(WHH), BT, 0, stream>>>(Wq, wq16, WHH);
  cvt_f16_kernel<<<blks(WHH), BT, 0, stream>>>(Wk, wk16, WHH);
  cvt_f16_kernel<<<blks(WHH), BT, 0, stream>>>(We, we16, WHH);
  cvt_f16_kernel<<<blks(E * G_H), BT, 0, stream>>>(edge_feats, e16, E * G_H);
  cvt_f16_kernel<<<blks(N * G_H), BT, 0, stream>>>(node_feats, h16, N * G_H);
  zero_u32_kernel<<<blks(N * G_H), BT, 0, stream>>>((unsigned*)pooled, N * G_H);
  zero_u32_kernel<<<blks(N), BT, 0, stream>>>(m_enc, N);
  zero_u32_kernel<<<blks(N), BT, 0, stream>>>((unsigned*)denom, N);
  zero_u32_kernel<<<1, G_H, 0, stream>>>((unsigned*)ctx, G_H);

  // ---- 8 GATv2 hops -------------------------------------------------------
  const unsigned gemm_blk = 128;                              // 4 waves
  const unsigned ngrid = (unsigned)((ntiles + 3) / 4);
  const unsigned egrid = (unsigned)((etiles + 3) / 4);

  for (int hop = 0; hop < G_HOPS; ++hop) {
    const _Float16* wq_h = wq16 + (size_t)hop * G_H * G_H;
    const _Float16* wk_h = wk16 + (size_t)hop * G_H * G_H;
    const _Float16* we_h = we16 + (size_t)hop * G_H * G_H;
    const float* b_h = bias + (size_t)hop * G_H;
    const float* a_h = att  + (size_t)hop * G_H;

    node_gemm_kernel<<<ngrid, gemm_blk, 0, stream>>>(
        h16, wq_h, wk_h, q, kn, ntiles, N);
    edge_logits_kernel<<<egrid, gemm_blk, 0, stream>>>(
        e16, we_h, senders, receivers, q, kn, b_h, a_h,
        v16, logits, m_enc, etiles, E);
    edge_exp_kernel<<<blks(E), BT, 0, stream>>>(
        logits, receivers, m_enc, ex, denom, E);
    edge_pool_kernel<<<blks(E * 16), BT, 0, stream>>>(
        ex, denom, receivers, v16, pooled, E);
    node_update_kernel<<<blks(N * G_H), BT, 0, stream>>>(
        pooled, h16, m_enc, denom, ctx, N, hop == G_HOPS - 1 ? 1 : 0);
  }

  finalize_kernel<<<1, G_H, 0, stream>>>(ctx, out, 1.0f / (float)N);
}
